// Net_17051020165226
// MI455X (gfx1250) — compile-verified, matrix operand order checked by reading, not run on hardware
//
#include <hip/hip_runtime.h>
#include <hip/hip_bf16.h>
#include <cmath>
#include <type_traits>

// Problem constants (match reference)
#define BB     128
#define LL     28
#define FFDIM  28
#define DMODEL 256
#define DINNER 512          // EXPAND * DM
#define DSTATE 16
#define DRANK  16           // DM/16
#define KCONV  3
#define NLAYER 5
#define OUTC   10
#define MROWS  (BB*LL)      // 3584 rows = 224 tiles of 16

typedef __attribute__((ext_vector_type(16))) _Float16 v16h;
typedef __attribute__((ext_vector_type(8)))  _Float16 v8h;
typedef __attribute__((ext_vector_type(8)))  float    v8f;

__device__ __forceinline__ float dev_sigmoid(float x) { return 1.0f / (1.0f + expf(-x)); }

// ---------------------------------------------------------------------------
// f16 fragment load for one 32-wide K chunk (CDNA5 16-bit 16x32 layout).
// p = row_ptr + kc + kbase, kbase = (lane>=16 ? 8 : 0). Elements 0..7 hold
// K=kbase..kbase+7 (one b128), elements 8..15 hold K=16+kbase.. (one b128).
// ---------------------------------------------------------------------------
__device__ __forceinline__ v16h load_frag16(const _Float16* __restrict__ p)
{
    v8h lo = *reinterpret_cast<const v8h*>(p);
    v8h hi = *reinterpret_cast<const v8h*>(p + 16);
    return __builtin_shufflevector(lo, hi,
                                   0, 1, 2, 3, 4, 5, 6, 7,
                                   8, 9, 10, 11, 12, 13, 14, 15);
}

// ---------------------------------------------------------------------------
// WMMA f16 GEMM: C[M,N] = A[M,K] * W[N,K]^T, K % 32 == 0, f32 accumulate.
// Each wave computes a 16 x (16*NSUB) tile: one A fragment feeds NSUB WMMAs
// (A-reuse x NSUB, 4 wmma per ~10 b128 loads at NSUB=4).
// EPI: 0 = store, 1 = softplus(acc + bias[n]), 2 = C += acc.
// ---------------------------------------------------------------------------
template <int EPI, int NSUB>
__global__ void gemm16_wmma(const _Float16* __restrict__ A, const _Float16* __restrict__ W,
                            float* __restrict__ C, const float* __restrict__ bias,
                            int K, int lda, int ldb, int ldc,
                            int groupsN, int nGroups)
{
    int grp = blockIdx.x * (blockDim.x >> 5) + (threadIdx.x >> 5);
    if (grp >= nGroups) return;                       // uniform per-wave exit
    int tm = grp / groupsN;
    int tg = grp % groupsN;
    int lane  = threadIdx.x & 31;
    int rsel  = lane & 15;                            // row (A) / col (B) in tile
    int kbase = (lane >> 4) << 3;                     // lanes 16-31: K-halves +8
    int n0    = tg * 16 * NSUB;

    const _Float16* pa = A + (size_t)(tm * 16 + rsel) * lda + kbase;
    const _Float16* pw = W + (size_t)(n0 + rsel) * ldb + kbase;

    v8f acc[NSUB];
#pragma unroll
    for (int s = 0; s < NSUB; ++s) acc[s] = (v8f){0.f,0.f,0.f,0.f,0.f,0.f,0.f,0.f};

    for (int kc = 0; kc < K; kc += 32) {
        __builtin_prefetch((const void*)(pa + kc + 128), 0, 0);  // global_prefetch_b8
        v16h a = load_frag16(pa + kc);
#pragma unroll
        for (int s = 0; s < NSUB; ++s) {
            v16h b = load_frag16(pw + (size_t)s * 16 * ldb + kc);
            acc[s] = __builtin_amdgcn_wmma_f32_16x16x32_f16(
                         /*neg_a=*/false, a, /*neg_b=*/false, b,
                         /*c_mod=*/(short)0, acc[s],
                         /*reuse_a=*/false, /*reuse_b=*/false);
        }
    }

    // C/D layout: VGPR r -> M = r + 8*(lane>=16), N = lane&15
    int mbase = tm * 16 + ((lane >> 4) << 3);
#pragma unroll
    for (int s = 0; s < NSUB; ++s) {
        int n = n0 + s * 16 + rsel;
#pragma unroll
        for (int r = 0; r < 8; ++r) {
            size_t off = (size_t)(mbase + r) * ldc + n;
            float v = acc[s][r];
            if (EPI == 1) {
                v += bias[n];
                C[off] = (v > 20.0f) ? v : log1pf(expf(v));   // softplus
            } else if (EPI == 2) {
                C[off] += v;                                   // residual accumulate
            } else {
                C[off] = v;
            }
        }
    }
}

// ---------------------------------------------------------------------------
// f32 -> f16 elementwise convert (weight / activation staging).
// ---------------------------------------------------------------------------
__global__ void cvt_f16_kernel(const float* __restrict__ src,
                               _Float16* __restrict__ dst, int n)
{
    int idx = blockIdx.x * blockDim.x + threadIdx.x;
    if (idx < n) dst[idx] = (_Float16)src[idx];
}

// ---------------------------------------------------------------------------
// Zero-pad a (rows x kin) f32 matrix (row stride ldin) into (rows x 32) f16.
// ---------------------------------------------------------------------------
__global__ void pad32_f16_kernel(const float* __restrict__ src,
                                 _Float16* __restrict__ dst,
                                 int rows, int kin, int ldin)
{
    int idx = blockIdx.x * blockDim.x + threadIdx.x;
    if (idx >= rows * 32) return;
    int k = idx & 31;
    int r = idx >> 5;
    dst[idx] = (k < kin) ? (_Float16)src[(size_t)r * ldin + k] : (_Float16)0.0f;
}

// ---------------------------------------------------------------------------
// Depthwise causal conv (K=3, left pad 2) + bias + SiLU -> f16 u.
// xz is (B*L, 2*DI) f32; channels [0,DI) feed the conv.
// ---------------------------------------------------------------------------
__global__ void conv_silu_kernel(const float* __restrict__ xz,
                                 const float* __restrict__ cw,
                                 const float* __restrict__ cb,
                                 _Float16* __restrict__ u16)
{
    int idx = blockIdx.x * blockDim.x + threadIdx.x;
    if (idx >= MROWS * DINNER) return;
    int d  = idx % DINNER;
    int bl = idx / DINNER;
    int l  = bl % LL;
    int b  = bl / LL;

    float acc = cb[d];
#pragma unroll
    for (int k = 0; k < KCONV; ++k) {
        int li = l + k - (KCONV - 1);
        if (li >= 0)
            acc += xz[((size_t)(b * LL + li)) * (2 * DINNER) + d] * cw[d * KCONV + k];
    }
    u16[idx] = (_Float16)(acc * dev_sigmoid(acc));    // SiLU
}

// ---------------------------------------------------------------------------
// Selective scan: one thread per (b, d); 16-wide state in registers,
// sequential over t. Fuses +u*D and the silu(z) gate; emits f16 y.
// dbc rows: [0:16)=dt, [16:32)=B, [32:48)=C.
// ---------------------------------------------------------------------------
__global__ void scan_kernel(const _Float16* __restrict__ u16,
                            const float* __restrict__ delta,
                            const float* __restrict__ dbc,
                            const float* __restrict__ xz,
                            const float* __restrict__ A_log,
                            const float* __restrict__ Dv,
                            _Float16* __restrict__ y16)
{
    int idx = blockIdx.x * blockDim.x + threadIdx.x;
    if (idx >= BB * DINNER) return;
    int d = idx % DINNER;
    int b = idx / DINNER;

    float a[DSTATE], st[DSTATE];
#pragma unroll
    for (int s = 0; s < DSTATE; ++s) {
        a[s]  = -expf(A_log[(size_t)d * DSTATE + s]);
        st[s] = 0.0f;
    }
    float Dd = Dv[d];

    for (int t = 0; t < LL; ++t) {
        size_t row = (size_t)(b * LL + t);
        float dl = delta[row * DINNER + d];
        float ut = (float)u16[row * DINNER + d];
        float du = dl * ut;
        const float* bc = dbc + row * 48;
        float yv = 0.0f;
#pragma unroll
        for (int s = 0; s < DSTATE; ++s) {
            st[s] = expf(dl * a[s]) * st[s] + du * bc[16 + s];
            yv += st[s] * bc[32 + s];
        }
        float zt = xz[row * (2 * DINNER) + DINNER + d];
        y16[row * DINNER + d] = (_Float16)((yv + ut * Dd) * (zt * dev_sigmoid(zt)));
    }
}

// ---------------------------------------------------------------------------
// Mean-pool over L, then tiny (B,256)x(256,10) classifier.
// ---------------------------------------------------------------------------
__global__ void pool_classify_kernel(const float* __restrict__ h,
                                     const float* __restrict__ cw,
                                     float* __restrict__ out)
{
    __shared__ float pooled[DMODEL];
    int b = blockIdx.x;
    for (int d = threadIdx.x; d < DMODEL; d += blockDim.x) {
        float s = 0.0f;
        for (int l = 0; l < LL; ++l)
            s += h[((size_t)(b * LL + l)) * DMODEL + d];
        pooled[d] = s * (1.0f / LL);
    }
    __syncthreads();
    if (threadIdx.x < OUTC) {
        float s = 0.0f;
        for (int d = 0; d < DMODEL; ++d)
            s += pooled[d] * cw[threadIdx.x * DMODEL + d];
        out[b * OUTC + threadIdx.x] = s;
    }
}

// ---------------------------------------------------------------------------
extern "C" void kernel_launch(void* const* d_in, const int* in_sizes, int n_in,
                              void* d_out, int out_size, void* d_ws, size_t ws_size,
                              hipStream_t stream)
{
    const float* x    = (const float*)d_in[0];   // (B,1,L,F)
    const float* win  = (const float*)d_in[1];   // (DM,F)
    const float* winp = (const float*)d_in[2];   // (NL,2*DI,DM)
    const float* cw   = (const float*)d_in[3];   // (NL,DI,K)
    const float* cb   = (const float*)d_in[4];   // (NL,DI)
    const float* xw   = (const float*)d_in[5];   // (NL,48,DI)
    const float* dtw  = (const float*)d_in[6];   // (NL,DI,DR)
    const float* dtb  = (const float*)d_in[7];   // (NL,DI)
    const float* alog = (const float*)d_in[8];   // (NL,DI,DS)
    const float* Dv   = (const float*)d_in[9];   // (NL,DI)
    const float* ow   = (const float*)d_in[10];  // (NL,DM,DI)
    const float* clw  = (const float*)d_in[11];  // (OUT,DM)
    float* out = (float*)d_out;

    // ---- Workspace layout: f32 region then f16 region (all 16B aligned) ----
    const size_t nH   = (size_t)MROWS * DMODEL;      // 917504
    const size_t nXZ  = (size_t)MROWS * 2 * DINNER;  // 3670016
    const size_t nU   = (size_t)MROWS * DINNER;      // 1835008
    const size_t nDBC = (size_t)MROWS * 48;          // 172032
    const size_t nXP  = (size_t)MROWS * 32;          // 114688
    const size_t nWIN = (size_t)DMODEL * 32;         // 8192
    const size_t nWDT = (size_t)DINNER * 32;         // 16384
    const size_t nWI  = (size_t)2 * DINNER * DMODEL; // 262144
    const size_t nWX  = (size_t)48 * DINNER;         // 24576
    const size_t nWO  = (size_t)DMODEL * DINNER;     // 131072

    const size_t f32elems = nH + nXZ + nDBC + nU;
    const size_t f16elems = nU + nU + nH + nXP + nWIN + nXP + nWDT + nWI + nWX + nWO;
    const size_t need = f32elems * sizeof(float) + f16elems * sizeof(_Float16);
    if (ws_size < need) return;

    float* h     = (float*)d_ws;
    float* xz    = h + nH;
    float* dbc   = xz + nXZ;
    float* delta = dbc + nDBC;

    _Float16* u16    = (_Float16*)(delta + nU);
    _Float16* y16    = u16 + nU;
    _Float16* h16    = y16 + nU;
    _Float16* xpad16 = h16 + nH;       // MROWS x 32
    _Float16* winp16 = xpad16 + nXP;   // DMODEL x 32 (padded input_proj_w)
    _Float16* dtA16  = winp16 + nWIN;  // MROWS x 32
    _Float16* dtW16  = dtA16 + nXP;    // DINNER x 32
    _Float16* wIn16  = dtW16 + nWDT;   // 1024 x 256
    _Float16* wX16   = wIn16 + nWI;    // 48 x 512
    _Float16* wO16   = wX16 + nWX;     // 256 x 512

    auto gemm = [&](auto epi_tag, auto nsub_tag,
                    const _Float16* A, const _Float16* W, float* C,
                    const float* bias, int M, int N, int K,
                    int lda, int ldb, int ldc) {
        constexpr int EPI  = decltype(epi_tag)::value;
        constexpr int NSUB = decltype(nsub_tag)::value;
        int groupsN = (N / 16) / NSUB;
        int nG = (M / 16) * groupsN;
        const int wpb = 4;                           // 4 waves per block
        dim3 blk(32 * wpb), grd((nG + wpb - 1) / wpb);
        gemm16_wmma<EPI, NSUB><<<grd, blk, 0, stream>>>(A, W, C, bias, K,
                                                        lda, ldb, ldc, groupsN, nG);
    };
    using E0 = std::integral_constant<int, 0>;
    using E1 = std::integral_constant<int, 1>;
    using E2 = std::integral_constant<int, 2>;
    using S3 = std::integral_constant<int, 3>;
    using S4 = std::integral_constant<int, 4>;

    auto cvt = [&](const float* src, _Float16* dst, size_t n) {
        cvt_f16_kernel<<<((int)n + 255) / 256, 256, 0, stream>>>(src, dst, (int)n);
    };
    auto pad = [&](const float* src, _Float16* dst, int rows, int kin, int ldin) {
        int n = rows * 32;
        pad32_f16_kernel<<<(n + 255) / 256, 256, 0, stream>>>(src, dst, rows, kin, ldin);
    };

    // h = x @ input_proj_w.T  (K=28 zero-padded to 32, staged f16)
    pad(x,   xpad16, MROWS,  FFDIM, FFDIM);
    pad(win, winp16, DMODEL, FFDIM, FFDIM);
    gemm(E0{}, S4{}, xpad16, winp16, h, nullptr, MROWS, DMODEL, 32, 32, 32, DMODEL);

    for (int i = 0; i < NLAYER; ++i) {
        // stage this layer's weights + current h in f16
        cvt(winp + (size_t)i * nWI, wIn16, nWI);
        cvt(xw   + (size_t)i * nWX, wX16,  nWX);
        cvt(ow   + (size_t)i * nWO, wO16,  nWO);
        cvt(h, h16, nH);

        // xz = h @ in_w.T    (M=3584, N=1024, K=256)
        gemm(E0{}, S4{}, h16, wIn16, xz, nullptr,
             MROWS, 2 * DINNER, DMODEL, DMODEL, DMODEL, 2 * DINNER);
        // u = silu(causal_dwconv(xb) + cb)  -> f16
        {
            int n = MROWS * DINNER;
            conv_silu_kernel<<<(n + 255) / 256, 256, 0, stream>>>(
                xz, cw + (size_t)i * DINNER * KCONV, cb + (size_t)i * DINNER, u16);
        }
        // dbc = u @ x_proj_w.T  (N=48 -> NSUB=3)
        gemm(E0{}, S3{}, u16, wX16, dbc, nullptr,
             MROWS, 48, DINNER, DINNER, DINNER, 48);
        // delta = softplus(dt @ dt_proj_w.T + dtb)  (K=16 padded to 32)
        pad(dbc, dtA16, MROWS, DRANK, 48);
        pad(dtw + (size_t)i * DINNER * DRANK, dtW16, DINNER, DRANK, DRANK);
        gemm(E1{}, S4{}, dtA16, dtW16, delta, dtb + (size_t)i * DINNER,
             MROWS, DINNER, 32, 32, 32, DINNER);
        // selective scan + gate -> f16 y
        {
            int n = BB * DINNER;
            scan_kernel<<<(n + 255) / 256, 256, 0, stream>>>(
                u16, delta, dbc, xz,
                alog + (size_t)i * DINNER * DSTATE, Dv + (size_t)i * DINNER, y16);
        }
        // h += y @ out_w.T
        gemm(E2{}, S4{}, y16, wO16, h, nullptr,
             MROWS, DMODEL, DINNER, DINNER, DINNER, DMODEL);
    }

    pool_classify_kernel<<<BB, 256, 0, stream>>>(h, clw, out);
}
